// INSTNCT_45638322487979
// MI455X (gfx1250) — compile-verified, matrix operand order checked by reading, not run on hardware
//
#include <hip/hip_runtime.h>
#include <hip/hip_bf16.h>

typedef __attribute__((ext_vector_type(16))) __bf16 v16bf;
typedef __attribute__((ext_vector_type(8)))  __bf16 v8bf;
typedef __attribute__((ext_vector_type(8)))  float  v8f;

#define M_RING   4096
#define D_DIM    256
#define N_EXP    8
#define R_RAD    4
#define F_DIM    64
#define BB_      16
#define T_STEPS  32
#define WSZ      9
#define S_SCALE  0.5f

// ---------------- zero workspace (ring + hidden state) ----------------
__global__ void INSTNCT_zero4(float4* __restrict__ p, long n4) {
    long i  = (long)blockIdx.x * blockDim.x + threadIdx.x;
    long st = (long)gridDim.x * blockDim.x;
    float4 z; z.x = 0.f; z.y = 0.f; z.z = 0.f; z.w = 0.f;
    for (; i < n4; i += st) p[i] = z;
}

// ---------------- pre-swizzle weights into WMMA B-fragment order ------
// B[k][n] = W[n][k].  Fragment element order matches the CDNA5 16-bit
// B-matrix (32x16) per-lane layout: lane L holds column n = L&15,
// kh = L>>4, elements e<8 -> k = ks*32 + kh*8 + e, e>=8 -> +16.
__global__ void INSTNCT_swizzle(const float* __restrict__ src,
                                __bf16* __restrict__ dst,
                                int nks, int kdim, int total) {
    int id = blockIdx.x * blockDim.x + threadIdx.x;
    if (id >= total) return;
    int e    = id & 15;
    int L    = (id >> 4) & 31;
    int tile = id >> 9;           // tile = nt*nks + ks
    int ks   = tile % nks;
    int nt   = tile / nks;
    int kh   = L >> 4;
    int k    = ks * 32 + kh * 8 + (e < 8 ? e : e + 8);
    int row  = nt * 16 + (L & 15);
    dst[id] = (__bf16)src[row * kdim + k];
}

// ---------------- A fragment loader from LDS (row-major [16][256]) ----
__device__ inline v16bf load_a_frag(const __bf16* A, int lane, int ks) {
    int r  = lane & 15;
    int kh = lane >> 4;
    const __bf16* p = A + r * D_DIM + ks * 32 + kh * 8;
    v8bf lo = *(const v8bf*)p;          // k .. k+7
    v8bf hi = *(const v8bf*)(p + 16);   // k+16 .. k+23
    v16bf a;
#pragma unroll
    for (int j = 0; j < 8; ++j) { a[j] = lo[j]; a[8 + j] = hi[j]; }
    return a;
}

// ---------------- persistent single-workgroup recurrent kernel --------
__global__ __launch_bounds__(512)
void INSTNCT_main(const float* __restrict__ x,
                  const float* __restrict__ b_in,
                  const float* __restrict__ b_out,
                  const float* __restrict__ b_read,
                  const int*   __restrict__ dests,
                  float*       __restrict__ ring,     // [16][4096][256]
                  float*       __restrict__ hidden,   // [8][16][256]
                  const __bf16* __restrict__ fragIn,  // 16nt x 2ks x 512
                  const __bf16* __restrict__ fragRead,// 8e x 16nt x 8ks x 512
                  const __bf16* __restrict__ fragOut, // 4nt x 8ks x 512
                  float*       __restrict__ out)      // [16][32][64]
{
    __shared__ __align__(16) float  inp_lds[BB_][D_DIM];   // 16 KB
    __shared__ __align__(16) float  hbuf[BB_][D_DIM];      // 16 KB
    __shared__ __align__(16) float  hsum[BB_][D_DIM];      // 16 KB
    __shared__ __align__(16) __bf16 A_lds[BB_][D_DIM];     //  8 KB
    __shared__ float ptr_lds[N_EXP][BB_];
    __shared__ int   center_lds[BB_];

    const int tid  = threadIdx.x;
    const int lane = tid & 31;
    const int wv   = tid >> 5;     // wave id 0..15

    if (tid < N_EXP * BB_) ptr_lds[tid >> 4][tid & 15] = 0.f;
    __syncthreads();

    for (int t = 0; t < T_STEPS; ++t) {
        // ---- stage x_t as bf16 A matrix; clear hsum ----
#pragma unroll
        for (int r = 0; r < 2; ++r) {
            int q = tid + r * 512;            // 1024 = 16*64 elements
            int b = q >> 6, f = q & 63;
            A_lds[b][f] = (__bf16)x[(b * T_STEPS + t) * F_DIM + f];
        }
#pragma unroll
        for (int r = 0; r < 8; ++r) {
            int q = tid + r * 512;            // 4096 elements
            hsum[q >> 8][q & 255] = 0.f;
        }
        __syncthreads();

        // ---- inp = x_t @ W_in^T + b_in : 16 waves, K=64 (2 wmma) ----
        {
            v8f acc = {0.f,0.f,0.f,0.f,0.f,0.f,0.f,0.f};
#pragma unroll
            for (int ks = 0; ks < 2; ++ks) {
                v16bf a  = load_a_frag(&A_lds[0][0], lane, ks);
                v16bf bm = *(const v16bf*)(fragIn + ((wv * 2 + ks) * 512 + lane * 16));
                acc = __builtin_amdgcn_wmma_f32_16x16x32_bf16(
                          false, a, false, bm, (short)0, acc, false, false);
            }
            int n = lane & 15, col = wv * 16 + n, mb = (lane >> 4) * 8;
#pragma unroll
            for (int v = 0; v < 8; ++v)
                inp_lds[mb + v][col] = acc[v] + b_in[col];
        }
        __syncthreads();

        // ---- sequential expert chain ----
        for (int i = 0; i < N_EXP; ++i) {
            if (lane == 0 && wv == 0 && i + 1 < N_EXP)
                __builtin_prefetch(fragRead + (size_t)(i + 1) * 65536, 0, 0);
            if (tid < BB_) {
                int c = (int)ptr_lds[i][tid];
                c = c < 0 ? 0 : (c > M_RING - 1 ? M_RING - 1 : c);
                center_lds[tid] = c;
            }
            __syncthreads();

            // gather window, average -> bf16 A; hpre = hidden+inp
            {
                int b  = tid >> 5;
                int db = (tid & 31) * 8;
                int c  = center_lds[b];
                float s[8];
#pragma unroll
                for (int j = 0; j < 8; ++j) s[j] = 0.f;
                const float* rb = ring + (size_t)b * M_RING * D_DIM;
#pragma unroll
                for (int w = 0; w < WSZ; ++w) {
                    int slot = (c + w - R_RAD) & (M_RING - 1);
                    const float4* p = (const float4*)(rb + slot * D_DIM + db);
                    float4 a0 = p[0], a1 = p[1];
                    s[0] += a0.x; s[1] += a0.y; s[2] += a0.z; s[3] += a0.w;
                    s[4] += a1.x; s[5] += a1.y; s[6] += a1.z; s[7] += a1.w;
                }
                const float inv = 1.0f / (float)WSZ;
                const float* hg = hidden + ((size_t)i * BB_ + b) * D_DIM + db;
#pragma unroll
                for (int j = 0; j < 8; ++j) {
                    A_lds[b][db + j] = (__bf16)(s[j] * inv);
                    hbuf[b][db + j]  = hg[j] + inp_lds[b][db + j];
                }
            }
            __syncthreads();

            // h += S * (read @ W_read[i]^T + b_read[i]) : K=256 (8 wmma)
            {
                v8f acc = {0.f,0.f,0.f,0.f,0.f,0.f,0.f,0.f};
                const __bf16* bf = fragRead + (size_t)i * 65536;
#pragma unroll
                for (int ks = 0; ks < 8; ++ks) {
                    v16bf a  = load_a_frag(&A_lds[0][0], lane, ks);
                    v16bf bm = *(const v16bf*)(bf + ((wv * 8 + ks) * 512 + lane * 16));
                    acc = __builtin_amdgcn_wmma_f32_16x16x32_bf16(
                              false, a, false, bm, (short)0, acc, false, false);
                }
                int n = lane & 15, col = wv * 16 + n, mb = (lane >> 4) * 8;
#pragma unroll
                for (int v = 0; v < 8; ++v) {
                    float hv = hbuf[mb + v][col] +
                               S_SCALE * (acc[v] + b_read[i * D_DIM + col]);
                    hbuf[mb + v][col] = hv;
                }
            }
            __syncthreads();

            // hidden writeback, hsum accum, race-free scatter-add
            {
                int b  = tid >> 5;
                int db = (tid & 31) * 8;
                int c  = center_lds[b];
                float* hg = hidden + ((size_t)i * BB_ + b) * D_DIM + db;
                float vloc[8];
#pragma unroll
                for (int j = 0; j < 8; ++j) {
                    float hv = hbuf[b][db + j];
                    hg[j] = hv;
                    hsum[b][db + j] += hv;
                    vloc[j] = hv * (1.0f / (float)WSZ);
                }
                float* rb = ring + (size_t)b * M_RING * D_DIM;
#pragma unroll
                for (int w = 0; w < WSZ; ++w) {
                    int slot = (c + w - R_RAD) & (M_RING - 1);
                    float4* p = (float4*)(rb + slot * D_DIM + db);
                    float4 a0 = p[0], a1 = p[1];
                    a0.x += vloc[0]; a0.y += vloc[1]; a0.z += vloc[2]; a0.w += vloc[3];
                    a1.x += vloc[4]; a1.y += vloc[5]; a1.z += vloc[6]; a1.w += vloc[7];
                    p[0] = a0; p[1] = a1;
                }
            }
            // pointer update
            if (tid < BB_) {
                float pr = ptr_lds[i][tid];
                int   c  = center_lds[tid];
                float jump = (float)dests[i * M_RING + c];
                float walk = pr + 1.0f;
                if (walk >= (float)M_RING) walk -= (float)M_RING;
                float p = (i % 3 == 0) ? 0.8f : ((i % 3 == 1) ? 0.2f : 0.5f);
                ptr_lds[i][tid] = p * jump + (1.0f - p) * walk;
            }
            __threadfence_block();
            __syncthreads();
        }

        // ---- out_t = mean(hidden) @ W_out^T + b_out ----
#pragma unroll
        for (int r = 0; r < 8; ++r) {
            int q = tid + r * 512;
            A_lds[q >> 8][q & 255] = (__bf16)(hsum[q >> 8][q & 255] * 0.125f);
        }
        __syncthreads();
        if (wv < 4) {
            v8f acc = {0.f,0.f,0.f,0.f,0.f,0.f,0.f,0.f};
#pragma unroll
            for (int ks = 0; ks < 8; ++ks) {
                v16bf a  = load_a_frag(&A_lds[0][0], lane, ks);
                v16bf bm = *(const v16bf*)(fragOut + ((wv * 8 + ks) * 512 + lane * 16));
                acc = __builtin_amdgcn_wmma_f32_16x16x32_bf16(
                          false, a, false, bm, (short)0, acc, false, false);
            }
            int n = lane & 15, col = wv * 16 + n, mb = (lane >> 4) * 8;
#pragma unroll
            for (int v = 0; v < 8; ++v)
                out[(mb + v) * (T_STEPS * F_DIM) + t * F_DIM + col] = acc[v] + b_out[col];
        }
        __syncthreads();
    }
}

// ---------------- launch ----------------
extern "C" void kernel_launch(void* const* d_in, const int* in_sizes, int n_in,
                              void* d_out, int out_size, void* d_ws, size_t ws_size,
                              hipStream_t stream) {
    const float* x      = (const float*)d_in[0];
    const float* W_in   = (const float*)d_in[1];
    const float* b_in   = (const float*)d_in[2];
    const float* W_out  = (const float*)d_in[3];
    const float* b_out  = (const float*)d_in[4];
    const float* W_read = (const float*)d_in[5];
    const float* b_read = (const float*)d_in[6];
    const int*   dests  = (const int*)d_in[7];

    char* ws = (char*)d_ws;
    const size_t RING_B   = (size_t)BB_ * M_RING * D_DIM * 4;   // 64 MB
    const size_t HID_B    = (size_t)N_EXP * BB_ * D_DIM * 4;    // 128 KB
    const size_t FRD_B    = (size_t)N_EXP * 16 * 8 * 512 * 2;   // 1 MB
    const size_t FIN_B    = (size_t)16 * 2 * 512 * 2;           // 32 KB
    float*  ring     = (float*)ws;
    float*  hidden   = (float*)(ws + RING_B);
    __bf16* fragRead = (__bf16*)(ws + RING_B + HID_B);
    __bf16* fragIn   = (__bf16*)(ws + RING_B + HID_B + FRD_B);
    __bf16* fragOut  = (__bf16*)(ws + RING_B + HID_B + FRD_B + FIN_B);

    // deterministic re-init of recurrent state every call
    long n4 = (long)((RING_B + HID_B) / 16);
    INSTNCT_zero4<<<2048, 256, 0, stream>>>((float4*)ws, n4);

    // pre-swizzle weights into WMMA B-fragment layout (bf16)
    for (int i = 0; i < N_EXP; ++i)
        INSTNCT_swizzle<<<(65536 + 255) / 256, 256, 0, stream>>>(
            W_read + (size_t)i * 65536, fragRead + (size_t)i * 65536, 8, 256, 65536);
    INSTNCT_swizzle<<<(16384 + 255) / 256, 256, 0, stream>>>(W_in,  fragIn,  2, 64,  16384);
    INSTNCT_swizzle<<<(16384 + 255) / 256, 256, 0, stream>>>(W_out, fragOut, 8, 256, 16384);

    // the whole recurrent chain in one persistent workgroup (16 wave32)
    INSTNCT_main<<<1, 512, 0, stream>>>(x, b_in, b_out, b_read, dests,
                                        ring, hidden, fragIn, fragRead, fragOut,
                                        (float*)d_out);
}
